// HybridVoxelNeuralField_58394375357325
// MI455X (gfx1250) — matrix operand order repaired and checked by reading, not compiled, not run on hardware
//
#include <hip/hip_runtime.h>
#include <hip/hip_bf16.h>
#include <cstddef>

typedef __attribute__((ext_vector_type(16))) _Float16 v16h;
typedef __attribute__((ext_vector_type(8)))  _Float16 v8h;
typedef __attribute__((ext_vector_type(8)))  float    v8f;

#define RES   128
#define VOX   (128ull*128ull*128ull)   // 2,097,152 voxels
#define NPTS  (2048*1024)              // 2,097,152 points

// ---------------------------------------------------------------------------
// Kernel 1: transpose grid (C,D,H,W) f32  ->  (D,H,W,C) f16 in workspace.
// Reads are coalesced per channel plane; each thread writes 64 contiguous B.
// ---------------------------------------------------------------------------
__global__ __launch_bounds__(256) void grid_transpose_f16(
    const float* __restrict__ g, _Float16* __restrict__ o) {
  const size_t idx = (size_t)blockIdx.x * 256 + threadIdx.x;   // voxel id
  v8h pk[4];
  #pragma unroll
  for (int c = 0; c < 32; ++c) {
    float v = g[(size_t)c * VOX + idx];
    pk[c >> 3][c & 7] = (_Float16)v;
  }
  v8h* dst = (v8h*)(o + idx * 32);
  #pragma unroll
  for (int q = 0; q < 4; ++q) dst[q] = pk[q];
}

// ---------------------------------------------------------------------------
// Kernel 2: fused trilinear sample + 4-layer MLP (WMMA f16 -> f32 accum).
// One point per lane for sampling; per wave, two 16-point M-tiles for WMMA.
// ---------------------------------------------------------------------------
template <bool TR>
__global__ __launch_bounds__(256) void voxel_mlp_fused(
    const float* __restrict__ coord,
    const float* __restrict__ gridF,      // channel-major f32 (fallback)
    const _Float16* __restrict__ gridH,   // voxel-major f16 (fast path)
    const float* __restrict__ w1, const float* __restrict__ b1,
    const float* __restrict__ w2, const float* __restrict__ b2,
    const float* __restrict__ w3, const float* __restrict__ b3,
    const float* __restrict__ w4, const float* __restrict__ b4,
    float* __restrict__ out) {

  __shared__ __align__(16) _Float16 ldsFeat[8 * 32 * 32];  // 16 KB
  __shared__ __align__(16) float    ldsOut [8 * 32 * 4];   //  4 KB

  const int tid  = threadIdx.x;
  const int lane = tid & 31;
  const int wav  = tid >> 5;
  const int col  = lane & 15;     // N column inside a 16-wide tile
  const int kh   = lane >> 4;     // which K-half / M-half this lane owns

  _Float16* slice  = ldsFeat + wav * (32 * 32);
  float*    oslice = ldsOut  + wav * (32 * 4);

  const size_t p = (size_t)blockIdx.x * 256 + tid;           // point id

  // ---------------- Phase A: trilinear gather (one point per lane) --------
  const float cx = coord[3 * p + 0];
  const float cy = coord[3 * p + 1];
  const float cz = coord[3 * p + 2];
  const float fx = ((cx + 1.f) * (float)RES - 1.f) * 0.5f;
  const float fy = ((cy + 1.f) * (float)RES - 1.f) * 0.5f;
  const float fz = ((cz + 1.f) * (float)RES - 1.f) * 0.5f;
  const float x0f = floorf(fx), y0f = floorf(fy), z0f = floorf(fz);
  const float tx = fx - x0f, ty = fy - y0f, tz = fz - z0f;
  const int x0 = (int)x0f, y0 = (int)y0f, z0 = (int)z0f;

  float acc[32];
  #pragma unroll
  for (int k = 0; k < 32; ++k) acc[k] = 0.f;

  #pragma unroll
  for (int dz = 0; dz < 2; ++dz) {
    #pragma unroll
    for (int dy = 0; dy < 2; ++dy) {
      #pragma unroll
      for (int dx = 0; dx < 2; ++dx) {
        const int xi = x0 + dx, yi = y0 + dy, zi = z0 + dz;
        const float w = (dz ? tz : 1.f - tz) * (dy ? ty : 1.f - ty) *
                        (dx ? tx : 1.f - tx);
        if ((unsigned)xi < (unsigned)RES && (unsigned)yi < (unsigned)RES &&
            (unsigned)zi < (unsigned)RES) {
          const size_t vox = ((size_t)zi * RES + yi) * RES + xi;
          if (TR) {
            const v8h* src = (const v8h*)(gridH + vox * 32);   // 64B, 1 line
            #pragma unroll
            for (int q = 0; q < 4; ++q) {
              v8h h = src[q];
              #pragma unroll
              for (int e = 0; e < 8; ++e) acc[q * 8 + e] += w * (float)h[e];
            }
          } else {
            #pragma unroll
            for (int c = 0; c < 32; ++c)
              acc[c] += w * gridF[(size_t)c * VOX + vox];
          }
        }
      }
    }
  }

  // Stage features to LDS as f16, point-major (32 f16 per point).
  {
    _Float16* dst = slice + lane * 32;
    #pragma unroll
    for (int q = 0; q < 4; ++q) {
      v8h t;
      #pragma unroll
      for (int e = 0; e < 8; ++e) t[e] = (_Float16)acc[q * 8 + e];
      ((v8h*)dst)[q] = t;
    }
  }
  __builtin_amdgcn_wave_barrier();

  // ---------------- Weight fragments (B layout, built once) ---------------
  // B (32x16 f16): column N=col; elements 0..7 are K = kh*8..kh*8+7,
  // elements 8..15 are K = 16+kh*8..16+kh*8+7 (same order as A), read from
  // weight row (out-feature) `nt*16+col`.
  auto loadB = [&](const float* W, int nt) -> v16h {
    v16h b;
    const float* wr = W + (size_t)(nt * 16 + col) * 32 + kh * 8;
    #pragma unroll
    for (int j = 0; j < 16; ++j) b[j] = (_Float16)wr[j < 8 ? j : j + 8];
    return b;
  };
  const v16h B1a = loadB(w1, 0), B1b = loadB(w1, 1);
  const v16h B2a = loadB(w2, 0), B2b = loadB(w2, 1);
  const v16h B3a = loadB(w3, 0), B3b = loadB(w3, 1);
  v16h B4;
  {
    #pragma unroll
    for (int j = 0; j < 16; ++j) B4[j] = (_Float16)0.f;
    if (col < 4) {
      const float* wr = w4 + (size_t)col * 32 + kh * 8;
      #pragma unroll
      for (int j = 0; j < 16; ++j) B4[j] = (_Float16)wr[j < 8 ? j : j + 8];
    }
  }
  const float bv1[2] = { b1[col], b1[16 + col] };
  const float bv2[2] = { b2[col], b2[16 + col] };
  const float bv3[2] = { b3[col], b3[16 + col] };
  const float bv4    = (col < 4) ? b4[col] : 0.f;

  // A fragment (16x32 f16): lane owns row M=lane%16; elements 0..7 are
  // K = kh*8..kh*8+7, elements 8..15 are K = 16+kh*8..16+kh*8+7.
  auto loadA = [&](int tb) -> v16h {
    union { v16h v; v8h h[2]; } u;
    const _Float16* base = slice + (size_t)(tb + col) * 32 + kh * 8;
    u.h[0] = *(const v8h*)(base);
    u.h[1] = *(const v8h*)(base + 16);
    return u.v;
  };
  // D tile (16x16 f32): VGPR r, lane -> element (M = r + 8*kh, N = col).
  auto storeRelu = [&](v8f d, int tb, int nt, float bias) {
    #pragma unroll
    for (int r = 0; r < 8; ++r) {
      float v = d[r] + bias;
      v = v > 0.f ? v : 0.f;
      slice[(size_t)(tb + r + 8 * kh) * 32 + nt * 16 + col] = (_Float16)v;
    }
  };

  // ---------------- Phase B: MLP, two 16-point tiles per wave -------------
  #pragma unroll
  for (int t = 0; t < 2; ++t) {
    const int tb = t * 16;
    v8f z = {};

    v16h A = loadA(tb);
    v8f d0 = __builtin_amdgcn_wmma_f32_16x16x32_f16(false, A, false, B1a, (short)0, z, false, false);
    v8f d1 = __builtin_amdgcn_wmma_f32_16x16x32_f16(false, A, false, B1b, (short)0, z, false, false);
    storeRelu(d0, tb, 0, bv1[0]); storeRelu(d1, tb, 1, bv1[1]);
    __builtin_amdgcn_wave_barrier();

    A  = loadA(tb);
    d0 = __builtin_amdgcn_wmma_f32_16x16x32_f16(false, A, false, B2a, (short)0, z, false, false);
    d1 = __builtin_amdgcn_wmma_f32_16x16x32_f16(false, A, false, B2b, (short)0, z, false, false);
    storeRelu(d0, tb, 0, bv2[0]); storeRelu(d1, tb, 1, bv2[1]);
    __builtin_amdgcn_wave_barrier();

    A  = loadA(tb);
    d0 = __builtin_amdgcn_wmma_f32_16x16x32_f16(false, A, false, B3a, (short)0, z, false, false);
    d1 = __builtin_amdgcn_wmma_f32_16x16x32_f16(false, A, false, B3b, (short)0, z, false, false);
    storeRelu(d0, tb, 0, bv3[0]); storeRelu(d1, tb, 1, bv3[1]);
    __builtin_amdgcn_wave_barrier();

    A  = loadA(tb);
    d0 = __builtin_amdgcn_wmma_f32_16x16x32_f16(false, A, false, B4, (short)0, z, false, false);
    if (col < 4) {
      #pragma unroll
      for (int r = 0; r < 8; ++r)
        oslice[(size_t)(tb + r + 8 * kh) * 4 + col] = d0[r] + bv4;
    }
    __builtin_amdgcn_wave_barrier();
  }

  // Coalesced output: one float4 per lane (== one point's 4 outputs).
  {
    const float4 v = ((const float4*)oslice)[lane];
    ((float4*)out)[p] = v;
  }
}

// ---------------------------------------------------------------------------
extern "C" void kernel_launch(void* const* d_in, const int* in_sizes, int n_in,
                              void* d_out, int out_size, void* d_ws, size_t ws_size,
                              hipStream_t stream) {
  const float* coord = (const float*)d_in[0];
  const float* grid  = (const float*)d_in[1];
  const float* w1 = (const float*)d_in[2];
  const float* b1 = (const float*)d_in[3];
  const float* w2 = (const float*)d_in[4];
  const float* b2 = (const float*)d_in[5];
  const float* w3 = (const float*)d_in[6];
  const float* b3 = (const float*)d_in[7];
  const float* w4 = (const float*)d_in[8];
  const float* b4 = (const float*)d_in[9];
  float* out = (float*)d_out;

  const size_t needT = VOX * 32ull * sizeof(_Float16);   // 128 MB
  const int blocks = (int)(NPTS / 256);                  // 8192, exact

  if (ws_size >= needT) {
    _Float16* gT = (_Float16*)d_ws;
    grid_transpose_f16<<<(int)(VOX / 256), 256, 0, stream>>>(grid, gT);
    voxel_mlp_fused<true><<<blocks, 256, 0, stream>>>(
        coord, grid, gT, w1, b1, w2, b2, w3, b3, w4, b4, out);
  } else {
    voxel_mlp_fused<false><<<blocks, 256, 0, stream>>>(
        coord, grid, (const _Float16*)nullptr, w1, b1, w2, b2, w3, b3, w4, b4, out);
  }
}
